// MultiHeadAttention_5634997092425
// MI455X (gfx1250) — matrix operand order repaired and checked by reading, not compiled, not run on hardware
//
#include <hip/hip_runtime.h>

typedef __bf16        v16bf __attribute__((ext_vector_type(16)));
typedef float         v8f   __attribute__((ext_vector_type(8)));
typedef unsigned int  u32x4 __attribute__((ext_vector_type(4)));

static constexpr int Bb = 4;
static constexpr int Tt = 2048;
static constexpr int Cc = 1024;
static constexpr int Hh = 16;
static constexpr int HS = 64;

union AFrag { v16bf v; u32x4 q[2]; };

#if __has_builtin(__builtin_amdgcn_global_load_async_to_lds_b128)
#define HAS_ASYNC_LDS 1
#else
#define HAS_ASYNC_LDS 0
#endif

// exact parameter types of the async builtin (from hipcc diagnostic):
// arg0: addrspace(1) pointer to int __vector(4); arg1: LDS pointer to same
typedef int i32v4 __attribute__((vector_size(16)));
typedef __attribute__((address_space(1))) i32v4* gvp_t;
typedef __attribute__((address_space(3))) i32v4* lvp_t;

// 16-byte global -> LDS copy; async (ASYNCcnt-tracked) when available.
__device__ __forceinline__ void copy16_g2l(const unsigned short* g, unsigned short* l) {
#if HAS_ASYNC_LDS
  __builtin_amdgcn_global_load_async_to_lds_b128(
      (gvp_t)(__attribute__((address_space(1))) void*)(void*)g,
      (lvp_t)(__attribute__((address_space(3))) void*)l, 0, 0);
#else
  *(u32x4*)l = *(const u32x4*)g;
#endif
}

template<int N>
__device__ __forceinline__ void wait_async() {
#if __has_builtin(__builtin_amdgcn_s_wait_asynccnt)
  __builtin_amdgcn_s_wait_asynccnt(N);
#else
  asm volatile("s_wait_asynccnt %0" :: "i"(N) : "memory");
#endif
}

__device__ __forceinline__ unsigned short f2bf(float f) {
  unsigned int u = __builtin_bit_cast(unsigned int, f);
  u += 0x7fffu + ((u >> 16) & 1u);           // round-to-nearest-even
  return (unsigned short)(u >> 16);
}

// ---------------- elementwise fp32 -> bf16 ----------------
__global__ void conv_f32_bf16(const float* __restrict__ src,
                              unsigned short* __restrict__ dst, int n) {
  int i = blockIdx.x * blockDim.x + threadIdx.x;
  if (i < n) dst[i] = f2bf(src[i]);
}

// W: (H, C, HS) fp32  ->  Wt: (N=H*HS rows, K=C cols) bf16   (Wt[n][k] = W[h][k][d])
__global__ void conv_wqkv(const float* __restrict__ W,
                          unsigned short* __restrict__ Wt) {
  int idx = blockIdx.x * blockDim.x + threadIdx.x;   // n*C + k
  if (idx >= Cc * Cc) return;
  int n = idx / Cc, k = idx % Cc;
  int hh = n / HS, d = n % HS;
  Wt[idx] = f2bf(W[((size_t)hh * Cc + k) * HS + d]);
}

// ------- bf16 WMMA GEMM: Y(MxN) = A(MxK) * Bw(NxK)^T (+bias) ----------------
// Block = 8 waves = 128x64 tile. Shared 64x32 B tile staged in LDS,
// double-buffered via async global->LDS copies (ASYNCcnt).
template<bool BIAS, bool OUTF32>
__global__ __launch_bounds__(256)
void gemm_bf16_wmma(const unsigned short* __restrict__ A,
                    const unsigned short* __restrict__ Bw,
                    const float* __restrict__ bias,
                    float* __restrict__ Yf,
                    unsigned short* __restrict__ Yb,
                    int M, int N, int K) {
  const int tid  = threadIdx.x;
  const int lane = tid & 31;
  const int wav  = tid >> 5;
  const int hl   = lane >> 4;       // which 16-lane half
  const int mn   = lane & 15;
  const int mbase = blockIdx.x * 128 + wav * 16;
  const int nbase = blockIdx.y * 64;

  __shared__ __align__(16) unsigned short Bs[2][64 * 32];   // [buf][row][k]

  // each of the 256 threads owns one 16B chunk of the 4KB B tile
  const int brow   = tid >> 2;      // 0..63
  const int bchunk = tid & 3;       // 0..3
  const unsigned short* gsrc = Bw + (size_t)(nbase + brow) * K + bchunk * 8;
  unsigned short* lb[2] = { &Bs[0][brow * 32 + bchunk * 8],
                            &Bs[1][brow * 32 + bchunk * 8] };

  v8f acc[4];
  #pragma unroll
  for (int i = 0; i < 4; ++i) acc[i] = (v8f)0.0f;

  copy16_g2l(gsrc, lb[0]);                         // prologue: k0 = 0 tile

  const unsigned short* arow = A + (size_t)(mbase + mn) * K;
  for (int k0 = 0; k0 < K; k0 += 32) {
    const int  cur  = (k0 >> 5) & 1;
    const bool more = (k0 + 32) < K;
    if (more) copy16_g2l(gsrc + k0 + 32, lb[cur ^ 1]);
#if HAS_ASYNC_LDS
    if (more) wait_async<1>(); else wait_async<0>();
#endif
    __syncthreads();                               // whole B tile resident

    AFrag af;   // 16-bit A 16x32 layout: chunks at elem off g*16 + hl*8
    af.q[0] = *(const u32x4*)(arow + k0 + hl * 8);
    af.q[1] = *(const u32x4*)(arow + k0 + 16 + hl * 8);
    #pragma unroll
    for (int nt = 0; nt < 4; ++nt) {
      const unsigned short* bsrow = &Bs[cur][(nt * 16 + mn) * 32];
      AFrag bfr;  // 16-bit B 32x16 layout: chunks at elem off g*16 + hl*8
      bfr.q[0] = *(const u32x4*)(bsrow + hl * 8);
      bfr.q[1] = *(const u32x4*)(bsrow + 16 + hl * 8);
      acc[nt] = __builtin_amdgcn_wmma_f32_16x16x32_bf16(
          false, af.v, false, bfr.v, (short)0, acc[nt], false, false);
    }
    __syncthreads();                               // done reading buf before overwrite
  }

  #pragma unroll
  for (int nt = 0; nt < 4; ++nt) {
    int col = nbase + nt * 16 + mn;
    float bv = BIAS ? bias[col] : 0.0f;
    #pragma unroll
    for (int r = 0; r < 8; ++r) {             // C/D layout: M = r + 8*hl, N = mn
      int row = mbase + r + 8 * hl;
      float v = acc[nt][r] + bv;
      if (OUTF32) Yf[(size_t)row * N + col] = v;
      else        Yb[(size_t)row * N + col] = f2bf(v);
    }
  }
}

// ---------------- causal flash attention (bf16 WMMA, fp32 accum) -----------
__global__ __launch_bounds__(128)
void flash_attn(const unsigned short* __restrict__ Q,
                const unsigned short* __restrict__ Kd,
                const unsigned short* __restrict__ V,
                unsigned short* __restrict__ O) {     // combo bf16 (B*T, C)
  const int tid  = threadIdx.x;
  const int lane = tid & 31;
  const int wav  = tid >> 5;
  const int hl   = lane >> 4;
  const int mn   = lane & 15;
  const int q0   = blockIdx.x * 64;
  const int qw   = q0 + wav * 16;
  const int h    = blockIdx.y;
  const int b    = blockIdx.z;
  const size_t base = ((size_t)b * Tt) * Cc + (size_t)h * HS;

  __shared__ __align__(16) unsigned short Vt[HS * 32];     // [d][key] (transposed)
  __shared__ __align__(16) unsigned short Ps[4][16 * 32];  // per-wave P [m][key]

  AFrag qa[2];
  {
    const unsigned short* qrow = Q + base + (size_t)(qw + mn) * Cc;
    #pragma unroll
    for (int c = 0; c < 2; ++c) {
      qa[c].q[0] = *(const u32x4*)(qrow + c * 32 + hl * 8);
      qa[c].q[1] = *(const u32x4*)(qrow + c * 32 + 16 + hl * 8);
    }
  }

  v8f o[4];
  #pragma unroll
  for (int i = 0; i < 4; ++i) o[i] = (v8f)0.0f;
  float mrow[8], lrow[8];
  #pragma unroll
  for (int r = 0; r < 8; ++r) { mrow[r] = -__builtin_inff(); lrow[r] = 0.0f; }

  const int kend = qw + 16;        // per-wave causal limit (exclusive)
  const int kall = q0 + 64;        // block-uniform loop bound

  for (int kb = 0; kb < kall; kb += 32) {
    // cooperative, transposed V tile load: 32 keys x 64 d
    for (int i = tid; i < HS * 32; i += 128) {
      int d = i & (HS - 1);
      int key = i >> 6;
      Vt[d * 32 + key] = V[base + (size_t)(kb + key) * Cc + d];
    }
    __syncthreads();

    if (kb < kend) {
      v8f s0 = (v8f)0.0f, s1 = (v8f)0.0f;
      #pragma unroll
      for (int c = 0; c < 2; ++c) {           // d-chunks of 32
        #pragma unroll
        for (int t = 0; t < 2; ++t) {         // key groups of 16
          const unsigned short* krow =
              Kd + base + (size_t)(kb + t * 16 + mn) * Cc + c * 32;
          AFrag kf;
          kf.q[0] = *(const u32x4*)(krow + hl * 8);
          kf.q[1] = *(const u32x4*)(krow + 16 + hl * 8);
          if (t == 0)
            s0 = __builtin_amdgcn_wmma_f32_16x16x32_bf16(
                false, qa[c].v, false, kf.v, (short)0, s0, false, false);
          else
            s1 = __builtin_amdgcn_wmma_f32_16x16x32_bf16(
                false, qa[c].v, false, kf.v, (short)0, s1, false, false);
        }
      }
      const float scale = 0.125f;             // HS^-0.5
      #pragma unroll
      for (int r = 0; r < 8; ++r) {
        int qidx = qw + r + 8 * hl;
        int key0 = kb + mn, key1 = kb + 16 + mn;
        float v0 = (key0 <= qidx) ? s0[r] * scale : -__builtin_inff();
        float v1 = (key1 <= qidx) ? s1[r] * scale : -__builtin_inff();
        float mx = fmaxf(v0, v1);
        #pragma unroll
        for (int msk = 1; msk < 16; msk <<= 1)
          mx = fmaxf(mx, __shfl_xor(mx, msk, 32));
        float mnew = fmaxf(mrow[r], mx);
        float corr = __expf(mrow[r] - mnew);
        float p0 = __expf(v0 - mnew);
        float p1 = __expf(v1 - mnew);
        float ps = p0 + p1;
        #pragma unroll
        for (int msk = 1; msk < 16; msk <<= 1) ps += __shfl_xor(ps, msk, 32);
        lrow[r] = lrow[r] * corr + ps;
        mrow[r] = mnew;
        #pragma unroll
        for (int nt = 0; nt < 4; ++nt) o[nt][r] *= corr;
        int m = r + 8 * hl;                   // C-layout -> A-layout via LDS
        Ps[wav][m * 32 + mn]      = f2bf(p0);
        Ps[wav][m * 32 + 16 + mn] = f2bf(p1);
      }
      // O += P (16x32) * V (32 x 64), V read transposed from LDS
      AFrag pa;
      const unsigned short* prow = &Ps[wav][mn * 32];
      pa.q[0] = *(const u32x4*)(prow + hl * 8);
      pa.q[1] = *(const u32x4*)(prow + 16 + hl * 8);
      #pragma unroll
      for (int nt = 0; nt < 4; ++nt) {
        const unsigned short* vrow = &Vt[(nt * 16 + mn) * 32];
        AFrag vf;
        vf.q[0] = *(const u32x4*)(vrow + hl * 8);
        vf.q[1] = *(const u32x4*)(vrow + 16 + hl * 8);
        o[nt] = __builtin_amdgcn_wmma_f32_16x16x32_bf16(
            false, pa.v, false, vf.v, (short)0, o[nt], false, false);
      }
    }
    __syncthreads();
  }

  #pragma unroll
  for (int r = 0; r < 8; ++r) {
    float inv = 1.0f / lrow[r];
    int row = b * Tt + qw + r + 8 * hl;
    #pragma unroll
    for (int nt = 0; nt < 4; ++nt)
      O[(size_t)row * Cc + h * HS + nt * 16 + mn] = f2bf(o[nt][r] * inv);
  }
}

// ------------------------------- host ---------------------------------------
extern "C" void kernel_launch(void* const* d_in, const int* in_sizes, int n_in,
                              void* d_out, int out_size, void* d_ws, size_t ws_size,
                              hipStream_t stream) {
  (void)in_sizes; (void)n_in; (void)out_size; (void)ws_size;
  const float* x     = (const float*)d_in[0];
  const float* Wq    = (const float*)d_in[1];
  const float* Wk    = (const float*)d_in[2];
  const float* Wv    = (const float*)d_in[3];
  const float* Wproj = (const float*)d_in[4];
  const float* bproj = (const float*)d_in[5];
  float* out = (float*)d_out;

  const int M = Bb * Tt;            // 8192
  const int N = Cc, K = Cc;
  const int BTC = Bb * Tt * Cc;     // 8388608
  const int CC  = Cc * Cc;          // 1048576

  char* ws = (char*)d_ws;
  size_t off = 0;
  auto carve = [&](size_t bytes) {
    void* p = ws + off;
    off += (bytes + 255) & ~(size_t)255;
    return p;
  };
  unsigned short* xb  = (unsigned short*)carve((size_t)BTC * 2);
  unsigned short* qb  = (unsigned short*)carve((size_t)BTC * 2);
  unsigned short* kb  = (unsigned short*)carve((size_t)BTC * 2);
  unsigned short* vb  = (unsigned short*)carve((size_t)BTC * 2);
  unsigned short* cb  = (unsigned short*)carve((size_t)BTC * 2);
  unsigned short* wqt = (unsigned short*)carve((size_t)CC * 2);
  unsigned short* wkt = (unsigned short*)carve((size_t)CC * 2);
  unsigned short* wvt = (unsigned short*)carve((size_t)CC * 2);
  unsigned short* wpt = (unsigned short*)carve((size_t)CC * 2);

  conv_f32_bf16<<<(BTC + 255) / 256, 256, 0, stream>>>(x, xb, BTC);
  conv_wqkv<<<(CC + 255) / 256, 256, 0, stream>>>(Wq, wqt);
  conv_wqkv<<<(CC + 255) / 256, 256, 0, stream>>>(Wk, wkt);
  conv_wqkv<<<(CC + 255) / 256, 256, 0, stream>>>(Wv, wvt);
  conv_f32_bf16<<<(CC + 255) / 256, 256, 0, stream>>>(Wproj, wpt, CC);

  dim3 ggrid(M / 128, N / 64);
  gemm_bf16_wmma<false, false><<<ggrid, 256, 0, stream>>>(xb, wqt, nullptr, nullptr, qb, M, N, K);
  gemm_bf16_wmma<false, false><<<ggrid, 256, 0, stream>>>(xb, wkt, nullptr, nullptr, kb, M, N, K);
  gemm_bf16_wmma<false, false><<<ggrid, 256, 0, stream>>>(xb, wvt, nullptr, nullptr, vb, M, N, K);

  flash_attn<<<dim3(Tt / 64, Hh, Bb), 128, 0, stream>>>(qb, kb, vb, cb);

  gemm_bf16_wmma<true, true><<<ggrid, 256, 0, stream>>>(cb, wpt, bproj, out, nullptr, M, N, K);
}
